// MultiHeadSelfAttention_5669356831057
// MI455X (gfx1250) — matrix operand order repaired and checked by reading, not compiled, not run on hardware
//
#include <hip/hip_runtime.h>

// ---------------------------------------------------------------------------
// MI455X / gfx1250 multi-head self-attention
//   B=4, S=2048, E=2048, H=16, D=128  (fp32 in/out, bf16 WMMA, TDM staging)
// ---------------------------------------------------------------------------

typedef __attribute__((ext_vector_type(8)))  float        v8f;
typedef __attribute__((ext_vector_type(16))) __bf16       v16bf;
typedef __attribute__((ext_vector_type(4)))  unsigned int u32x4;
typedef __attribute__((ext_vector_type(8)))  int          i32x8;
typedef __attribute__((ext_vector_type(4)))  int          i32x4;

union FragU {
    uint4 u[2];   // 32 bytes
    v16bf v;
};

static __device__ __forceinline__ unsigned short f2bf(float f) {
    unsigned u = __builtin_bit_cast(unsigned, f);
    u += 0x7FFFu + ((u >> 16) & 1u);        // round-to-nearest-even
    return (unsigned short)(u >> 16);
}
static __device__ __forceinline__ unsigned pack2(float a, float b) {
    return (unsigned)f2bf(a) | ((unsigned)f2bf(b) << 16);
}
static __device__ __forceinline__ v8f wmma_bf16(const FragU& a, const FragU& b, v8f c) {
    return __builtin_amdgcn_wmma_f32_16x16x32_bf16(false, a.v, false, b.v,
                                                   (short)0, c, false, false);
}
static __device__ __forceinline__ unsigned lds_off(const void* p) {
    // LDS aperture: generic address low 32 bits == byte offset in wave's LDS
    return (unsigned)(reinterpret_cast<uintptr_t>(p));
}

// ---------------------------------------------------------------------------
// TDM: 2D tile (tile0 elems/row x tile1 rows, 2-byte elems) Global -> LDS,
// row-padded in LDS. pad_interval: pad every 2^(v+1) DWORDs; pad_amount: v+1 DW.
// D# packing per CDNA5 ISA ch.8.3/8.4. 6-arg builtin (clang-23 arity).
// ---------------------------------------------------------------------------
static __device__ __forceinline__ void tdm_load_2d(unsigned lds_byte,
                                                   unsigned long long gaddr,
                                                   unsigned tile0, unsigned tile1,
                                                   unsigned long long stride0,
                                                   unsigned pad_interval,
                                                   unsigned pad_amount)
{
    const unsigned dim0 = 1u << 30, dim1 = 1u << 30;   // generous OOB bounds
    u32x4 g0;
    g0[0] = 1u;                                        // count=1 (valid), no gather
    g0[1] = lds_byte;                                  // lds_addr [63:32]
    g0[2] = (unsigned)(gaddr & 0xFFFFFFFFu);           // global_addr [95:64]
    g0[3] = (unsigned)((gaddr >> 32) & 0x01FFFFFFu) | (2u << 30);  // +type=2
    i32x8 g1;
    g1[0] = (int)((1u << 16) |                         // data_size = 2 bytes
                  (1u << 20) |                         // pad_enable
                  (pad_interval << 22) | (pad_amount << 25));
    g1[1] = (int)((dim0 & 0xFFFFu) << 16);             // tensor_dim0 lo
    g1[2] = (int)(((dim0 >> 16) & 0xFFFFu) | ((dim1 & 0xFFFFu) << 16));
    g1[3] = (int)(((dim1 >> 16) & 0xFFFFu) | (tile0 << 16));       // tile_dim0
    g1[4] = (int)(tile1 & 0xFFFFu);                    // tile_dim1, tile_dim2=0
    g1[5] = (int)(unsigned)(stride0 & 0xFFFFFFFFu);    // tensor_dim0_stride
    g1[6] = (int)(unsigned)((stride0 >> 32) & 0xFFFFu);
    g1[7] = 0;
    i32x4 z4; z4[0] = 0; z4[1] = 0; z4[2] = 0; z4[3] = 0;  // 2D: groups 2/3 unused
    i32x8 z8;
    #pragma unroll
    for (int i = 0; i < 8; ++i) z8[i] = 0;
    __builtin_amdgcn_tensor_load_to_lds(g0, g1, z4, z4, z8, 0);
}

// ---------------------------------------------------------------------------
// fp32 -> bf16 conversion, 8 elements / thread
// ---------------------------------------------------------------------------
__global__ __launch_bounds__(256) void cvt_kernel(const float* __restrict__ in,
                                                  unsigned short* __restrict__ out,
                                                  int n8)
{
    const int i = blockIdx.x * 256 + threadIdx.x;
    if (i >= n8) return;
    const float4* p = reinterpret_cast<const float4*>(in) + (size_t)i * 2;
    float4 a = p[0], b = p[1];
    uint4 o;
    o.x = pack2(a.x, a.y); o.y = pack2(a.z, a.w);
    o.z = pack2(b.x, b.y); o.w = pack2(b.z, b.w);
    reinterpret_cast<uint4*>(out)[i] = o;
}

// ---------------------------------------------------------------------------
// GEMM:  C[8192][2048] = A(bf16, MxK rm) * W(bf16, NxK rm)^T + bias(f32)
// EPI 0: bf16 [B,H,S,D]; EPI 1: bf16 [B,H,D,S]; EPI 2: f32 [M,N]
// 128x128 block tile, BK=32, 8 waves x (64x32). TDM double-buffered staging.
// ---------------------------------------------------------------------------
template <int EPI>
__global__ __launch_bounds__(256) void gemm_kernel(const unsigned short* __restrict__ A,
                                                   const unsigned short* __restrict__ W,
                                                   const float* __restrict__ bias,
                                                   void* __restrict__ outp)
{
    constexpr int LDSS = 40;                      // 32 halves + 8 pad
    __shared__ __align__(16) unsigned short sA[2][128 * LDSS];
    __shared__ __align__(16) unsigned short sW[2][128 * LDSS];

    const int t    = threadIdx.x;
    const int lane = t & 31;
    const int w    = t >> 5;
    const int n0   = blockIdx.x * 128;
    const int m0   = blockIdx.y * 128;
    const int mbase = (w >> 2) * 64;
    const int nbase = (w & 3) * 32;
    const int fn = lane & 15;
    const int fh = lane >> 4;

    v8f vz;
    #pragma unroll
    for (int e = 0; e < 8; ++e) vz[e] = 0.0f;
    v8f acc[4][2];
    #pragma unroll
    for (int i = 0; i < 4; ++i) {
        #pragma unroll
        for (int j = 0; j < 2; ++j) acc[i][j] = vz;
    }

    const unsigned long long abase = (unsigned long long)(uintptr_t)A + (size_t)m0 * 2048 * 2;
    const unsigned long long wbase = (unsigned long long)(uintptr_t)W + (size_t)n0 * 2048 * 2;

    if (w == 0) {   // prefetch chunk 0 (TDM ignores EXEC; one issue per wave)
        tdm_load_2d(lds_off(&sA[0][0]), abase, 32, 128, 2048, 3, 3);
        tdm_load_2d(lds_off(&sW[0][0]), wbase, 32, 128, 2048, 3, 3);
    }

    for (int kc = 0; kc < 64; ++kc) {
        if (w == 0) {
            if (kc + 1 < 64) {  // prefetch next chunk into the other buffer
                const unsigned long long koff = (unsigned long long)(kc + 1) * 32 * 2;
                const int nb = (kc + 1) & 1;
                tdm_load_2d(lds_off(&sA[nb][0]), abase + koff, 32, 128, 2048, 3, 3);
                tdm_load_2d(lds_off(&sW[nb][0]), wbase + koff, 32, 128, 2048, 3, 3);
                __builtin_amdgcn_s_wait_tensorcnt(2);   // chunk kc complete
            } else {
                __builtin_amdgcn_s_wait_tensorcnt(0);
            }
        }
        __syncthreads();
        const unsigned short* cA = &sA[kc & 1][0];
        const unsigned short* cW = &sW[kc & 1][0];

        FragU bfrag[2];
        #pragma unroll
        for (int nt = 0; nt < 2; ++nt) {
            const unsigned short* p = &cW[(nbase + nt * 16 + fn) * LDSS + fh * 16];
            bfrag[nt].u[0] = *reinterpret_cast<const uint4*>(p);
            bfrag[nt].u[1] = *reinterpret_cast<const uint4*>(p + 8);
        }
        #pragma unroll
        for (int mt = 0; mt < 4; ++mt) {
            FragU af;
            const unsigned short* ar = &cA[(mbase + mt * 16 + fn) * LDSS + fh * 16];
            af.u[0] = *reinterpret_cast<const uint4*>(ar);
            af.u[1] = *reinterpret_cast<const uint4*>(ar + 8);
            acc[mt][0] = wmma_bf16(af, bfrag[0], acc[mt][0]);
            acc[mt][1] = wmma_bf16(af, bfrag[1], acc[mt][1]);
        }
        __syncthreads();
    }

    #pragma unroll
    for (int mt = 0; mt < 4; ++mt) {
        #pragma unroll
        for (int nt = 0; nt < 2; ++nt) {
            const int gn = n0 + nbase + nt * 16 + fn;
            const float bv = bias[gn];
            #pragma unroll
            for (int r = 0; r < 8; ++r) {
                const int gm = m0 + mbase + mt * 16 + r + 8 * fh;
                const float val = acc[mt][nt][r] + bv;
                if constexpr (EPI == 0) {        // bf16 [B,H,S,D]
                    const int bb = gm >> 11, s = gm & 2047;
                    const int hh = gn >> 7,  d = gn & 127;
                    ((unsigned short*)outp)[(((size_t)(bb * 16 + hh) * 2048 + s) * 128) + d] = f2bf(val);
                } else if constexpr (EPI == 1) { // bf16 [B,H,D,S]
                    const int bb = gm >> 11, s = gm & 2047;
                    const int hh = gn >> 7,  d = gn & 127;
                    ((unsigned short*)outp)[((size_t)(bb * 16 + hh) * 128 + d) * 2048 + s] = f2bf(val);
                } else {                          // f32 [M,2048]
                    ((float*)outp)[(size_t)gm * 2048 + gn] = val;
                }
            }
        }
    }
}

// ---------------------------------------------------------------------------
// Flash attention. Block = one (b,h) x 128 q rows (8 waves x 16).
// 32-key chunks TDM-staged (double buffered). Causal+padding mask, online
// softmax. Output bf16 [B,S,H*D] feeding the bf16 output-projection GEMM.
// ---------------------------------------------------------------------------
__global__ __launch_bounds__(256) void attn_kernel(const unsigned short* __restrict__ Q,
                                                   const unsigned short* __restrict__ Kb,
                                                   const unsigned short* __restrict__ Vt,
                                                   const int* __restrict__ am,
                                                   unsigned short* __restrict__ Ob)
{
    constexpr int KST = 136;   // 128 + 8 pad (halves)
    constexpr int VST = 40;    // 32 + 8 pad
    constexpr int PST = 40;
    constexpr float SCALE = 0.08838834764831845f;   // 1/sqrt(128)
    __shared__ __align__(16) unsigned short sK[2][32 * KST];
    __shared__ __align__(16) unsigned short sV[2][128 * VST];
    __shared__ __align__(16) unsigned short sP[8 * 16 * PST];

    const int t    = threadIdx.x;
    const int lane = t & 31;
    const int w    = t >> 5;
    const int bh   = blockIdx.y;        // b*16 + h
    const int bidx = bh >> 4;
    const int head = bh & 15;
    const int qblk = blockIdx.x * 128;
    const int qb   = qblk + w * 16;
    const int fn = lane & 15, fh = lane >> 4;

    // Q fragments: 16x128 -> 4 A-fragments of 16x32 (registers)
    FragU qf[4];
    {
        const unsigned short* qrow = Q + ((size_t)bh * 2048 + (qb + fn)) * 128;
        #pragma unroll
        for (int f = 0; f < 4; ++f) {
            const unsigned short* p = qrow + f * 32 + fh * 16;
            qf[f].u[0] = *reinterpret_cast<const uint4*>(p);
            qf[f].u[1] = *reinterpret_cast<const uint4*>(p + 8);
        }
    }

    v8f vz;
    #pragma unroll
    for (int e = 0; e < 8; ++e) vz[e] = 0.0f;
    v8f oacc[8];
    #pragma unroll
    for (int dt = 0; dt < 8; ++dt) oacc[dt] = vz;
    float mrow[8], lrow[8];
    #pragma unroll
    for (int r = 0; r < 8; ++r) { mrow[r] = -3.0e38f; lrow[r] = 0.0f; }

    const unsigned long long kbbase = (unsigned long long)(uintptr_t)Kb +
                                      (size_t)bh * 2048 * 128 * 2;
    const unsigned long long vtbase = (unsigned long long)(uintptr_t)Vt +
                                      (size_t)bh * 128 * 2048 * 2;
    const int nchunks = (qblk + 128) / 32;

    if (w == 0) {   // prefetch chunk 0: K tile 32x128, Vt tile 128x32
        tdm_load_2d(lds_off(&sK[0][0]), kbbase, 128, 32, 128, 5, 3);
        tdm_load_2d(lds_off(&sV[0][0]), vtbase, 32, 128, 2048, 3, 3);
    }

    for (int kc = 0; kc < nchunks; ++kc) {
        const int kb = kc * 32;
        if (w == 0) {
            if (kc + 1 < nchunks) {
                const int nb = (kc + 1) & 1;
                tdm_load_2d(lds_off(&sK[nb][0]), kbbase + (size_t)(kb + 32) * 128 * 2,
                            128, 32, 128, 5, 3);
                tdm_load_2d(lds_off(&sV[nb][0]), vtbase + (size_t)(kb + 32) * 2,
                            32, 128, 2048, 3, 3);
                __builtin_amdgcn_s_wait_tensorcnt(2);
            } else {
                __builtin_amdgcn_s_wait_tensorcnt(0);
            }
        }
        __syncthreads();

        if (kb <= qb + 15) {            // else chunk fully causally masked
            const unsigned short* cK = &sK[kc & 1][0];
            const unsigned short* cV = &sV[kc & 1][0];

            // ---- scores: 2 tiles 16q x 16k, K-dim 128 (4 wmma each)
            v8f sc0 = vz, sc1 = vz;
            #pragma unroll
            for (int ks = 0; ks < 4; ++ks) {
                FragU b0, b1;
                const unsigned short* p0 = &cK[fn * KST + ks * 32 + fh * 16];
                b0.u[0] = *reinterpret_cast<const uint4*>(p0);
                b0.u[1] = *reinterpret_cast<const uint4*>(p0 + 8);
                const unsigned short* p1 = &cK[(16 + fn) * KST + ks * 32 + fh * 16];
                b1.u[0] = *reinterpret_cast<const uint4*>(p1);
                b1.u[1] = *reinterpret_cast<const uint4*>(p1 + 8);
                sc0 = wmma_bf16(qf[ks], b0, sc0);
                sc1 = wmma_bf16(qf[ks], b1, sc1);
            }

            // ---- mask + online softmax (row = r + 8*fh, col = fn)
            const int k0g = kb + fn, k1g = kb + 16 + fn;
            const int mv0 = am[bidx * 2048 + k0g];
            const int mv1 = am[bidx * 2048 + k1g];
            float p0v[8], p1v[8];
            #pragma unroll
            for (int r = 0; r < 8; ++r) {
                const int qg = qb + r + 8 * fh;
                float s0 = sc0[r] * SCALE;
                float s1 = sc1[r] * SCALE;
                if ((k0g > qg) || (mv0 == 0)) s0 = -3.0e38f;
                if ((k1g > qg) || (mv1 == 0)) s1 = -3.0e38f;
                float mx = fmaxf(s0, s1);
                #pragma unroll
                for (int off = 1; off < 16; off <<= 1)
                    mx = fmaxf(mx, __shfl_xor(mx, off, 32));
                const float mnew  = fmaxf(mrow[r], mx);
                const float alpha = __expf(mrow[r] - mnew);
                mrow[r] = mnew;
                s0 = (s0 > -1.0e37f) ? __expf(s0 - mnew) : 0.0f;
                s1 = (s1 > -1.0e37f) ? __expf(s1 - mnew) : 0.0f;
                float rs = s0 + s1;
                #pragma unroll
                for (int off = 1; off < 16; off <<= 1)
                    rs += __shfl_xor(rs, off, 32);
                lrow[r] = lrow[r] * alpha + rs;
                p0v[r] = s0; p1v[r] = s1;
                #pragma unroll
                for (int dt = 0; dt < 8; ++dt) oacc[dt][r] *= alpha;
            }

            // ---- C-layout -> A-layout for P via per-wave LDS tile
            unsigned short* pw = &sP[w * 16 * PST];
            #pragma unroll
            for (int r = 0; r < 8; ++r) {
                const int row = r + 8 * fh;
                pw[row * PST + fn]      = f2bf(p0v[r]);
                pw[row * PST + 16 + fn] = f2bf(p1v[r]);
            }
            FragU pf;
            {
                const unsigned short* pp = &pw[fn * PST + fh * 16];
                pf.u[0] = *reinterpret_cast<const uint4*>(pp);
                pf.u[1] = *reinterpret_cast<const uint4*>(pp + 8);
            }
            // ---- O += P @ V  (8 d-tiles, K-dim 32 keys)
            #pragma unroll
            for (int dt = 0; dt < 8; ++dt) {
                FragU vf;
                const unsigned short* pv = &cV[(dt * 16 + fn) * VST + fh * 16];
                vf.u[0] = *reinterpret_cast<const uint4*>(pv);
                vf.u[1] = *reinterpret_cast<const uint4*>(pv + 8);
                oacc[dt] = wmma_bf16(pf, vf, oacc[dt]);
            }
        }
        __syncthreads();
    }

    // ---- normalize and store bf16 into [B, S, H*D]
    float inv[8];
    #pragma unroll
    for (int r = 0; r < 8; ++r) inv[r] = lrow[r] > 0.0f ? 1.0f / lrow[r] : 0.0f;
    #pragma unroll
    for (int dt = 0; dt < 8; ++dt) {
        const int d = dt * 16 + fn;
        #pragma unroll
        for (int r = 0; r < 8; ++r) {
            const int qg = qb + r + 8 * fh;
            Ob[((size_t)bidx * 2048 + qg) * 2048 + head * 128 + d] =
                f2bf(oacc[dt][r] * inv[r]);
        }
    }
}

// ---------------------------------------------------------------------------
extern "C" void kernel_launch(void* const* d_in, const int* in_sizes, int n_in,
                              void* d_out, int out_size, void* d_ws, size_t ws_size,
                              hipStream_t stream) {
    (void)in_sizes; (void)n_in; (void)out_size; (void)ws_size;
    const float* x  = (const float*)d_in[0];
    const int*   am = (const int*)d_in[1];
    const float* Wq = (const float*)d_in[2];
    const float* bq = (const float*)d_in[3];
    const float* Wk = (const float*)d_in[4];
    const float* bk = (const float*)d_in[5];
    const float* Wv = (const float*)d_in[6];
    const float* bv = (const float*)d_in[7];
    const float* Wo = (const float*)d_in[8];
    const float* bo = (const float*)d_in[9];

    const size_t XE = (size_t)4 * 2048 * 2048;   // 16,777,216
    const size_t WE = (size_t)2048 * 2048;       //  4,194,304
    unsigned short* xbf = (unsigned short*)d_ws;
    unsigned short* Wqb = xbf + XE;
    unsigned short* Wkb = Wqb + WE;
    unsigned short* Wvb = Wkb + WE;
    unsigned short* Wob = Wvb + WE;
    unsigned short* Qbf = Wob + WE;
    unsigned short* Kbf = Qbf + XE;
    unsigned short* Vtb = Kbf + XE;
    unsigned short* aOb = Vtb + XE;              // ~192 MiB total workspace

    cvt_kernel<<<(int)(XE / 8 / 256), 256, 0, stream>>>(x,  xbf, (int)(XE / 8));
    cvt_kernel<<<(int)(WE / 8 / 256), 256, 0, stream>>>(Wq, Wqb, (int)(WE / 8));
    cvt_kernel<<<(int)(WE / 8 / 256), 256, 0, stream>>>(Wk, Wkb, (int)(WE / 8));
    cvt_kernel<<<(int)(WE / 8 / 256), 256, 0, stream>>>(Wv, Wvb, (int)(WE / 8));
    cvt_kernel<<<(int)(WE / 8 / 256), 256, 0, stream>>>(Wo, Wob, (int)(WE / 8));

    dim3 gg(16, 64);
    gemm_kernel<0><<<gg, 256, 0, stream>>>(xbf, Wqb, bq, (void*)Qbf);
    gemm_kernel<0><<<gg, 256, 0, stream>>>(xbf, Wkb, bk, (void*)Kbf);
    gemm_kernel<1><<<gg, 256, 0, stream>>>(xbf, Wvb, bv, (void*)Vtb);
    attn_kernel<<<dim3(16, 64), 256, 0, stream>>>(Qbf, Kbf, Vtb, am, aOb);
    gemm_kernel<2><<<gg, 256, 0, stream>>>(aOb, Wob, bo, (void*)d_out);
}
